// Point_Transformer_partseg_19765439497090
// MI455X (gfx1250) — matrix operand order repaired and checked by reading, not compiled, not run on hardware
//
#include <hip/hip_runtime.h>
#include <hip/hip_bf16.h>

typedef __attribute__((ext_vector_type(16))) _Float16 v16h;
typedef __attribute__((ext_vector_type(8)))  float    v8f;

#define CDIV(a, b) (((a) + (b) - 1) / (b))

// ---------------------------------------------------------------------------
// Generic WMMA GEMM:  C[M,L] = A(MxK) * X(KxL) (+bias[m]), batched over z.
//   A: row-major [M][lda]   (TRANS_A=false)  or  A[m,k] = Aptr[k*lda + m] (true)
//   X: row-major [K][L] (channel-major activations)
//   C: row-major [M][L]
// f32 inputs converted on the fly to f16 fragments (v_fma_mix cvt), f32
// accumulate via v_wmma_f32_16x16x32_f16. Each wave owns a 16x16 C tile;
// a 256-thread block (8 waves) covers a 32x64 tile.
// Main K-loop is branch-free (unconditional loads); only the K-tail uses
// branchless clamp+select.
// ---------------------------------------------------------------------------
template <bool TRANS_A>
__global__ __launch_bounds__(256)
void gemm_wmma_k(const float* __restrict__ A, long aBatch, int lda,
                 const float* __restrict__ X, long xBatch,
                 float* __restrict__ C, long cBatch,
                 const float* __restrict__ bias,
                 int M, int K, int L)
{
    const int bz = blockIdx.z;
    A += (long)bz * aBatch;
    X += (long)bz * xBatch;
    C += (long)bz * cBatch;

    const int wave = threadIdx.x >> 5;
    const int lane = threadIdx.x & 31;
    const int tm = (blockIdx.y * 2 + (wave >> 2)) * 16;
    const int tl = (blockIdx.x * 4 + (wave & 3)) * 16;
    if (tm >= M || tl >= L) return;   // wave-uniform: EXEC stays all-ones for WMMA

    const int half = lane >> 4;       // 0: lanes 0-15, 1: lanes 16-31
    const int l15  = lane & 15;

    // A fragment: lane holds row m = tm + l15; 16 halves cover a K-chunk of 32.
    const int   arow  = tm + l15;
    const float amask = (arow < M) ? 1.f : 0.f;
    const int   ar    = (arow < M) ? arow : (M - 1);
    const int   abase = half * 8;     // lanes16-31 start at K+8 (ISA 16-bit A layout)

    // B fragment: lane holds col n = tl + l15; same K order as A (lanes16-31 start at K+8).
    const int   bcol  = tl + l15;
    const float bmask = (bcol < L) ? 1.f : 0.f;
    const int   bcc   = (bcol < L) ? bcol : (L - 1);
    const int   bbase = half * 8;

    v8f acc = {0.f, 0.f, 0.f, 0.f, 0.f, 0.f, 0.f, 0.f};

    // One branch-free K-chunk of 32 (no bounds checks; caller guarantees k0+32<=K).
    auto chunk = [&](int k0) {
        v16h af, bf;
        if (TRANS_A) {
#pragma unroll
            for (int i = 0; i < 16; ++i) {
                int ka = k0 + ((i < 8) ? 0 : 16) + abase + (i & 7);
                af[i] = (_Float16)(A[(long)ka * lda + ar] * amask);
            }
        } else {
            const float* ap = A + (long)ar * lda + k0 + abase;
#pragma unroll
            for (int i = 0; i < 8; ++i) af[i]     = (_Float16)(ap[i]      * amask);
#pragma unroll
            for (int i = 0; i < 8; ++i) af[8 + i] = (_Float16)(ap[16 + i] * amask);
        }
        const float* xp = X + (long)(k0 + bbase) * L + bcc;
#pragma unroll
        for (int i = 0; i < 16; ++i)
            bf[i] = (_Float16)(xp[(long)(i + ((i < 8) ? 0 : 8)) * L] * bmask);
        acc = __builtin_amdgcn_wmma_f32_16x16x32_f16(false, af, false, bf,
                                                     (short)0, acc, false, false);
    };

    // Guarded K-tail: unconditional loads from clamped addresses + select-zero.
    auto chunk_tail = [&](int k0) {
        v16h af, bf;
#pragma unroll
        for (int i = 0; i < 16; ++i) {
            int ka = k0 + ((i < 8) ? 0 : 16) + abase + (i & 7);
            int kc = (ka < K) ? ka : (K - 1);
            float av = TRANS_A ? A[(long)kc * lda + ar] : A[(long)ar * lda + kc];
            av = (ka < K) ? av : 0.f;
            af[i] = (_Float16)(av * amask);

            int kb  = k0 + bbase + i + ((i < 8) ? 0 : 8);
            int kbc = (kb < K) ? kb : (K - 1);
            float bv = X[(long)kbc * L + bcc];
            bv = (kb < K) ? bv : 0.f;
            bf[i] = (_Float16)(bv * bmask);
        }
        acc = __builtin_amdgcn_wmma_f32_16x16x32_f16(false, af, false, bf,
                                                     (short)0, acc, false, false);
    };

    int k0 = 0;
    for (; k0 + 64 <= K; k0 += 64) {   // 2x WMMA per trip: amortize loop overhead
        chunk(k0);
        chunk(k0 + 32);
    }
    if (k0 + 32 <= K) { chunk(k0); k0 += 32; }
    if (k0 < K)       { chunk_tail(k0); }

    // C layout: VGPR i -> row tm + i (lanes 0-15) / tm + 8 + i (lanes 16-31)
    const int crow0 = tm + half * 8;
    if (bcol < L) {
#pragma unroll
        for (int i = 0; i < 8; ++i) {
            int cr = crow0 + i;
            if (cr < M) {
                float vv = acc[i];
                if (bias) vv += bias[cr];
                C[(long)cr * L + bcol] = vv;
            }
        }
    }
}

static void launch_gemm(const float* A, long aBatch, int lda,
                        const float* X, long xBatch,
                        float* C, long cBatch, const float* bias,
                        int M, int K, int L, int batch, bool transA,
                        hipStream_t s)
{
    dim3 grid(CDIV(L, 64), CDIV(M, 32), batch);
    if (transA)
        gemm_wmma_k<true ><<<grid, 256, 0, s>>>(A, aBatch, lda, X, xBatch, C, cBatch, bias, M, K, L);
    else
        gemm_wmma_k<false><<<grid, 256, 0, s>>>(A, aBatch, lda, X, xBatch, C, cBatch, bias, M, K, L);
}

// ---------------------------------------------------------------------------
// Training-mode BatchNorm: stats per channel over (batch, spatial).
// x layout: [Bc][C][NB] contiguous (Bc=1 for the [C][L] grouped tensors).
// ---------------------------------------------------------------------------
__global__ void bn_stats_k(const float* __restrict__ x, int C, int NB, int Bc,
                           float* __restrict__ mean, float* __restrict__ inv)
{
    const int c = blockIdx.x;
    const int t = threadIdx.x;
    __shared__ float s1[256], s2[256];
    float su = 0.f, sq = 0.f;
    const long total = (long)Bc * NB;
    for (long i = t; i < total; i += 256) {
        long b = i / NB, n = i % NB;
        float v = x[((long)b * C + c) * NB + n];
        su += v; sq += v * v;
    }
    s1[t] = su; s2[t] = sq;
    __syncthreads();
    for (int st = 128; st > 0; st >>= 1) {
        if (t < st) { s1[t] += s1[t + st]; s2[t] += s2[t + st]; }
        __syncthreads();
    }
    if (t == 0) {
        float m = s1[0] / (float)total;
        float var = s2[0] / (float)total - m * m;
        mean[c] = m;
        inv[c]  = rsqrtf(fmaxf(var, 0.f) + 1e-5f);
    }
}

__global__ void bn_apply_k(float* __restrict__ x, const float* __restrict__ mean,
                           const float* __restrict__ inv, const float* __restrict__ g,
                           const float* __restrict__ bb, int C, int NB, long total, int act)
{
    long idx = (long)blockIdx.x * blockDim.x + threadIdx.x;
    if (idx >= total) return;
    int c = (int)((idx / NB) % C);
    float v = (x[idx] - mean[c]) * inv[c] * g[c] + bb[c];
    if (act == 1)      v = fmaxf(v, 0.f);
    else if (act == 2) v = (v >= 0.f) ? v : 0.2f * v;
    x[idx] = v;
}

// ---------------------------------------------------------------------------
// Farthest point sampling: one block per batch; dist[] in LDS; sequential
// argmax with first-index tie-break (matches jnp.argmax).
// ---------------------------------------------------------------------------
__global__ void fps_k(const float* __restrict__ xyz, int N, int npoint, int* __restrict__ fidx)
{
    const int b = blockIdx.x;
    const int t = threadIdx.x;
    __shared__ float dist[2048];
    __shared__ float rv[256];
    __shared__ int   ri[256];
    __shared__ int   s_far;
    for (int i = t; i < N; i += 256) dist[i] = 1e10f;
    if (t == 0) s_far = 0;
    __syncthreads();
    for (int it = 0; it < npoint; ++it) {
        int far = s_far;
        if (t == 0) fidx[(long)b * npoint + it] = far;
        float cx = xyz[((long)b * N + far) * 3 + 0];
        float cy = xyz[((long)b * N + far) * 3 + 1];
        float cz = xyz[((long)b * N + far) * 3 + 2];
        float best = -1.f; int bi = 0x7fffffff;
        for (int i = t; i < N; i += 256) {
            float dx = xyz[((long)b * N + i) * 3 + 0] - cx;
            float dy = xyz[((long)b * N + i) * 3 + 1] - cy;
            float dz = xyz[((long)b * N + i) * 3 + 2] - cz;
            float nd = fminf(dist[i], dx * dx + dy * dy + dz * dz);
            dist[i] = nd;
            if (nd > best) { best = nd; bi = i; }
        }
        rv[t] = best; ri[t] = bi;
        __syncthreads();
        for (int st = 128; st > 0; st >>= 1) {
            if (t < st) {
                if (rv[t + st] > rv[t] || (rv[t + st] == rv[t] && ri[t + st] < ri[t])) {
                    rv[t] = rv[t + st]; ri[t] = ri[t + st];
                }
            }
            __syncthreads();
        }
        if (t == 0) s_far = ri[0];
        __syncthreads();
    }
}

// ---------------------------------------------------------------------------
// KNN: per (s,b) block computes N squared distances into LDS, then extracts
// the Kn smallest by iterative argmin (set is order-irrelevant: downstream is
// max-pool and BN over the K axis).
// ---------------------------------------------------------------------------
__global__ void knn_k(const float* __restrict__ q, const float* __restrict__ xyz,
                      int N, int S, int* __restrict__ knn, int Kn)
{
    const int s = blockIdx.x, b = blockIdx.y, t = threadIdx.x;
    __shared__ float d[2048];
    __shared__ float rv[256];
    __shared__ int   ri[256];
    float qx = q[((long)b * S + s) * 3 + 0];
    float qy = q[((long)b * S + s) * 3 + 1];
    float qz = q[((long)b * S + s) * 3 + 2];
    for (int i = t; i < N; i += 256) {
        float dx = xyz[((long)b * N + i) * 3 + 0] - qx;
        float dy = xyz[((long)b * N + i) * 3 + 1] - qy;
        float dz = xyz[((long)b * N + i) * 3 + 2] - qz;
        d[i] = dx * dx + dy * dy + dz * dz;
    }
    __syncthreads();
    for (int kk = 0; kk < Kn; ++kk) {
        float best = 3.0e38f; int bi = 0x7fffffff;
        for (int i = t; i < N; i += 256) {
            float v = d[i];
            if (v < best) { best = v; bi = i; }
        }
        rv[t] = best; ri[t] = bi;
        __syncthreads();
        for (int st = 128; st > 0; st >>= 1) {
            if (t < st) {
                if (rv[t + st] < rv[t] || (rv[t + st] == rv[t] && ri[t + st] < ri[t])) {
                    rv[t] = rv[t + st]; ri[t] = ri[t + st];
                }
            }
            __syncthreads();
        }
        if (t == 0) { knn[((long)b * S + s) * Kn + kk] = ri[0]; d[ri[0]] = 3.0e38f; }
        __syncthreads();
    }
}

// Build grouped features X[2D][B*S*Kn]: rows 0..D-1 = grouped - center,
// rows D..2D-1 = broadcast center. pts is channel-major [B][D][N].
__global__ void build_group_k(const float* __restrict__ pts, const int* __restrict__ fidx,
                              const int* __restrict__ knn, float* __restrict__ Xg,
                              int D, int N, int S, int Kn, int B)
{
    const long Lg = (long)B * S * Kn;
    const long total = Lg * D;
    long idx = (long)blockIdx.x * blockDim.x + threadIdx.x;
    if (idx >= total) return;
    long l = idx % Lg;
    int  c = (int)(idx / Lg);
    int  kk = (int)(l % Kn);
    int  s  = (int)((l / Kn) % S);
    int  b  = (int)(l / ((long)Kn * S));
    int  nn = knn[((long)b * S + s) * Kn + kk];
    int  fc = fidx[b * S + s];
    float pv = pts[((long)b * D + c) * N + nn];
    float cv = pts[((long)b * D + c) * N + fc];
    Xg[(long)c * Lg + l]       = pv - cv;
    Xg[(long)(D + c) * Lg + l] = cv;
}

// Max over the K axis of Y[C][B*S*Kn] -> f [B][C][S]
__global__ void group_max_k(const float* __restrict__ Y, float* __restrict__ f,
                            int C, int S, int Kn, int B)
{
    long total = (long)B * C * S;
    long idx = (long)blockIdx.x * blockDim.x + threadIdx.x;
    if (idx >= total) return;
    int s = (int)(idx % S);
    int c = (int)((idx / S) % C);
    int b = (int)(idx / ((long)S * C));
    long Lg = (long)B * S * Kn;
    long base = ((long)b * S + s) * Kn;
    float m = -3.0e38f;
    for (int kk = 0; kk < Kn; ++kk) m = fmaxf(m, Y[(long)c * Lg + base + kk]);
    f[((long)b * C + c) * S + s] = m;
}

// x [B][3][N] -> xyz [B][N][3]
__global__ void x_to_xyz_k(const float* __restrict__ x, float* __restrict__ xyz, int N, int B)
{
    long total = (long)B * N * 3;
    long idx = (long)blockIdx.x * blockDim.x + threadIdx.x;
    if (idx >= total) return;
    int k = (int)(idx % 3);
    int n = (int)((idx / 3) % N);
    int b = (int)(idx / ((long)3 * N));
    xyz[idx] = x[((long)b * 3 + k) * N + n];
}

// in [B][S][3] -> out [B][3][S]
__global__ void transpose3_k(const float* __restrict__ in, float* __restrict__ out, int S, int B)
{
    long total = (long)B * 3 * S;
    long idx = (long)blockIdx.x * blockDim.x + threadIdx.x;
    if (idx >= total) return;
    int n = (int)(idx % S);
    int k = (int)((idx / S) % 3);
    int b = (int)(idx / ((long)3 * S));
    out[idx] = in[((long)b * S + n) * 3 + k];
}

// dst[b][s][:] = src[b][fidx[b][s]][:]
__global__ void gather3_k(const float* __restrict__ src, const int* __restrict__ fidx,
                          float* __restrict__ dst, int N, int S, int B)
{
    long total = (long)B * S * 3;
    long idx = (long)blockIdx.x * blockDim.x + threadIdx.x;
    if (idx >= total) return;
    int k = (int)(idx % 3);
    int s = (int)((idx / 3) % S);
    int b = (int)(idx / ((long)3 * S));
    dst[idx] = src[((long)b * N + fidx[b * S + s]) * 3 + k];
}

__global__ void add_k(const float* __restrict__ a, const float* __restrict__ b,
                      float* __restrict__ o, long n)
{
    long i = (long)blockIdx.x * blockDim.x + threadIdx.x;
    if (i < n) o[i] = a[i] + b[i];
}

__global__ void sub_k(const float* __restrict__ a, const float* __restrict__ b,
                      float* __restrict__ o, long n)
{
    long i = (long)blockIdx.x * blockDim.x + threadIdx.x;
    if (i < n) o[i] = a[i] - b[i];
}

// Row softmax over last axis; one block per (b,n) row; att is L2-resident (67MB < 192MB)
__global__ void softmax_row_k(float* __restrict__ att, int N)
{
    long row = blockIdx.x;
    float* p = att + row * (long)N;
    __shared__ float red[256];
    __shared__ float bc;
    int t = threadIdx.x;
    float mx = -3.0e38f;
    for (int i = t; i < N; i += 256) mx = fmaxf(mx, p[i]);
    red[t] = mx;
    __syncthreads();
    for (int st = 128; st > 0; st >>= 1) {
        if (t < st) red[t] = fmaxf(red[t], red[t + st]);
        __syncthreads();
    }
    if (t == 0) bc = red[0];
    __syncthreads();
    float m = bc, sum = 0.f;
    for (int i = t; i < N; i += 256) { float e = expf(p[i] - m); p[i] = e; sum += e; }
    __syncthreads();
    red[t] = sum;
    __syncthreads();
    for (int st = 128; st > 0; st >>= 1) {
        if (t < st) red[t] += red[t + st];
        __syncthreads();
    }
    if (t == 0) bc = red[0];
    __syncthreads();
    float invs = 1.f / bc;
    for (int i = t; i < N; i += 256) p[i] *= invs;
}

// cs[b][m] = sum_n att[b][n][m]
__global__ void colsum_k(const float* __restrict__ att, float* __restrict__ cs, int N, int B)
{
    long total = (long)B * N;
    long idx = (long)blockIdx.x * blockDim.x + threadIdx.x;
    if (idx >= total) return;
    int b = (int)(idx / N), m = (int)(idx % N);
    const float* p = att + (long)b * N * N;
    float s = 0.f;
    for (int n = 0; n < N; ++n) s += p[(long)n * N + m];
    cs[idx] = s;
}

// x[b][c][m] /= (1e-9 + cs[b][m])  -- folds the offset-attention L1 renorm
__global__ void colscale_k(float* __restrict__ x, const float* __restrict__ cs,
                           int C, int N, long total)
{
    long idx = (long)blockIdx.x * blockDim.x + threadIdx.x;
    if (idx >= total) return;
    int m = (int)(idx % N);
    int b = (int)(idx / ((long)C * N));
    x[idx] = x[idx] / (1e-9f + cs[b * N + m]);
}

__global__ void concat4_k(const float* __restrict__ s0, const float* __restrict__ s1,
                          const float* __restrict__ s2, const float* __restrict__ s3,
                          float* __restrict__ o, int Cp, int N, long total)
{
    long idx = (long)blockIdx.x * blockDim.x + threadIdx.x;
    if (idx >= total) return;
    int n  = (int)(idx % N);
    int cc = (int)((idx / N) % (4 * Cp));
    int b  = (int)(idx / ((long)4 * Cp * N));
    const float* src = (cc < Cp) ? s0 : (cc < 2 * Cp) ? s1 : (cc < 3 * Cp) ? s2 : s3;
    int c = cc % Cp;
    o[idx] = src[((long)b * Cp + c) * N + n];
}

// xmax/xavg over N per (b,c)
__global__ void maxavg_k(const float* __restrict__ xf, float* __restrict__ xmax_,
                         float* __restrict__ xavg_, int N)
{
    int bc_ = blockIdx.x;
    const float* p = xf + (long)bc_ * N;
    __shared__ float rm[256], rs[256];
    int t = threadIdx.x;
    float m = -3.0e38f, s = 0.f;
    for (int i = t; i < N; i += 256) { float v = p[i]; m = fmaxf(m, v); s += v; }
    rm[t] = m; rs[t] = s;
    __syncthreads();
    for (int st = 128; st > 0; st >>= 1) {
        if (t < st) { rm[t] = fmaxf(rm[t], rm[t + st]); rs[t] += rs[t + st]; }
        __syncthreads();
    }
    if (t == 0) { xmax_[bc_] = rm[0]; xavg_[bc_] = rs[0] / (float)N; }
}

// y [B][3136][N]: [xf(1024) | xmax bcast(1024) | xavg bcast(1024) | lf bcast(64)]
__global__ void build_y_k(const float* __restrict__ xf, const float* __restrict__ xm,
                          const float* __restrict__ xa, const float* __restrict__ lf,
                          float* __restrict__ y, int N, long total)
{
    long idx = (long)blockIdx.x * blockDim.x + threadIdx.x;
    if (idx >= total) return;
    int n = (int)(idx % N);
    int c = (int)((idx / N) % 3136);
    int b = (int)(idx / ((long)3136 * N));
    float v;
    if (c < 1024)      v = xf[((long)b * 1024 + c) * N + n];
    else if (c < 2048) v = xm[b * 1024 + (c - 1024)];
    else if (c < 3072) v = xa[b * 1024 + (c - 2048)];
    else               v = lf[b * 64 + (c - 3072)];
    y[idx] = v;
}

// ---------------------------------------------------------------------------
extern "C" void kernel_launch(void* const* d_in, const int* in_sizes, int n_in,
                              void* d_out, int out_size, void* d_ws, size_t ws_size,
                              hipStream_t stream)
{
    (void)in_sizes; (void)n_in; (void)out_size;

    const int B = 4, N = 2048, S = 2048, KN = 32;
    const long Lg = (long)B * S * KN;          // 262144 grouped columns

    // ---- inputs (setup_inputs dict order) ----
    const float* x       = (const float*)d_in[0];
    const float* clsl    = (const float*)d_in[1];
    const float* conv1_w = (const float*)d_in[2];
    const float* bn1_g   = (const float*)d_in[3];
    const float* bn1_b   = (const float*)d_in[4];
    const float* conv2_w = (const float*)d_in[5];
    const float* bn2_g   = (const float*)d_in[6];
    const float* bn2_b   = (const float*)d_in[7];
    const float* pos_w   = (const float*)d_in[8];
    const float* pos_b   = (const float*)d_in[9];
    const float* l0_w1   = (const float*)d_in[10];
    const float* l0_g1   = (const float*)d_in[11];
    const float* l0_b1   = (const float*)d_in[12];
    const float* l0_w2   = (const float*)d_in[13];
    const float* l0_g2   = (const float*)d_in[14];
    const float* l0_b2   = (const float*)d_in[15];
    const float* l1_w1   = (const float*)d_in[16];
    const float* l1_g1   = (const float*)d_in[17];
    const float* l1_b1   = (const float*)d_in[18];
    const float* l1_w2   = (const float*)d_in[19];
    const float* l1_g2   = (const float*)d_in[20];
    const float* l1_b2   = (const float*)d_in[21];
    const float* int_w   = (const float*)d_in[22];
    const float* int_g   = (const float*)d_in[23];
    const float* int_b   = (const float*)d_in[24];
    const float* fuse_w  = (const float*)d_in[25];
    const float* fuse_g  = (const float*)d_in[26];
    const float* fuse_b  = (const float*)d_in[27];
    const float* lab_w   = (const float*)d_in[28];
    const float* lab_g   = (const float*)d_in[29];
    const float* lab_b   = (const float*)d_in[30];
    const float* s1_w    = (const float*)d_in[31];
    const float* s1_bias = (const float*)d_in[32];
    const float* s1_g    = (const float*)d_in[33];
    const float* s1_b    = (const float*)d_in[34];
    const float* s2_w    = (const float*)d_in[35];
    const float* s2_bias = (const float*)d_in[36];
    const float* s2_g    = (const float*)d_in[37];
    const float* s2_b    = (const float*)d_in[38];
    const float* s3_w    = (const float*)d_in[39];
    const float* s3_bias = (const float*)d_in[40];
    const float *wqk[4], *wv[4], *bv[4], *wt[4], *bt[4], *sag[4], *sab[4];
    for (int i = 0; i < 4; ++i) {
        int base = 41 + i * 7;
        wqk[i] = (const float*)d_in[base + 0];
        wv[i]  = (const float*)d_in[base + 1];
        bv[i]  = (const float*)d_in[base + 2];
        wt[i]  = (const float*)d_in[base + 3];
        bt[i]  = (const float*)d_in[base + 4];
        sag[i] = (const float*)d_in[base + 5];
        sab[i] = (const float*)d_in[base + 6];
    }

    // ---- workspace layout (bump allocator, deterministic every call) ----
    char* wsp = (char*)d_ws;
    size_t off = 0;
    auto alloc = [&](size_t bytes) -> void* {
        void* p = wsp + off;
        off += (bytes + 255) & ~(size_t)255;
        return p;
    };
    float* xyz   = (float*)alloc((size_t)B * N * 3 * 4);
    float* h1    = (float*)alloc((size_t)B * 64 * N * 4);
    float* h2    = (float*)alloc((size_t)B * 64 * N * 4);
    int*   fidx1 = (int*)  alloc((size_t)B * S * 4);
    int*   fidx2 = (int*)  alloc((size_t)B * S * 4);
    float* nx1   = (float*)alloc((size_t)B * S * 3 * 4);
    float* nx2   = (float*)alloc((size_t)B * S * 3 * 4);
    float* nx2T  = (float*)alloc((size_t)B * 3 * S * 4);
    int*   knn1  = (int*)  alloc((size_t)B * S * KN * 4);
    int*   knn2  = (int*)  alloc((size_t)B * S * KN * 4);
    float* f0    = (float*)alloc((size_t)B * 128 * N * 4);
    float* f1    = (float*)alloc((size_t)B * 256 * N * 4);
    float* pos   = (float*)alloc((size_t)B * 128 * N * 4);
    float* cur   = (float*)alloc((size_t)B * 128 * N * 4);
    float* xin   = (float*)alloc((size_t)B * 128 * N * 4);
    float* qb    = (float*)alloc((size_t)B * 32 * N * 4);
    float* vb    = (float*)alloc((size_t)B * 128 * N * 4);
    float* xr    = (float*)alloc((size_t)B * 128 * N * 4);
    float* td    = (float*)alloc((size_t)B * 128 * N * 4);
    float* ub    = (float*)alloc((size_t)B * 128 * N * 4);
    float* cs    = (float*)alloc((size_t)B * N * 4);
    float* saout = (float*)alloc((size_t)4 * B * 128 * N * 4);
    float* xc    = (float*)alloc((size_t)B * 512 * N * 4);
    float* xf    = (float*)alloc((size_t)B * 1024 * N * 4);
    float* xmaxb = (float*)alloc((size_t)B * 1024 * 4);
    float* xavgb = (float*)alloc((size_t)B * 1024 * 4);
    float* lf    = (float*)alloc((size_t)B * 64 * 4);
    float* y1    = (float*)alloc((size_t)B * 512 * N * 4);
    float* y2    = (float*)alloc((size_t)B * 256 * N * 4);
    float* meanb = (float*)alloc((size_t)4096 * 4);
    float* invb  = (float*)alloc((size_t)4096 * 4);
    float* bigA  = (float*)alloc((size_t)256 * Lg * 4);  // Xg / att / y3136
    float* bigB  = (float*)alloc((size_t)256 * Lg * 4);  // Yg
    if (off > ws_size) return;  // insufficient workspace: do nothing (safe)

    auto bn = [&](float* t, const float* g, const float* bb, int C, int NB, int Bc, int act) {
        bn_stats_k<<<C, 256, 0, stream>>>(t, C, NB, Bc, meanb, invb);
        long total = (long)Bc * C * NB;
        bn_apply_k<<<(unsigned)CDIV(total, 256), 256, 0, stream>>>(t, meanb, invb, g, bb, C, NB, total, act);
    };

    const long EN = (long)B * 128 * N;  // common elementwise size

    // ================= stem =================
    x_to_xyz_k<<<CDIV((long)B * N * 3, 256), 256, 0, stream>>>(x, xyz, N, B);
    launch_gemm(conv1_w, 0, 3,  x,  (long)3 * N,  h1, (long)64 * N, nullptr, 64, 3,  N, B, false, stream);
    bn(h1, bn1_g, bn1_b, 64, N, B, 1);
    launch_gemm(conv2_w, 0, 64, h1, (long)64 * N, h2, (long)64 * N, nullptr, 64, 64, N, B, false, stream);
    bn(h2, bn2_g, bn2_b, 64, N, B, 1);

    // ============ sample_and_group + local_op #0 (D=64 -> 128ch) ============
    fps_k<<<B, 256, 0, stream>>>(xyz, N, S, fidx1);
    gather3_k<<<CDIV((long)B * S * 3, 256), 256, 0, stream>>>(xyz, fidx1, nx1, N, S, B);
    knn_k<<<dim3(S, B), 256, 0, stream>>>(nx1, xyz, N, S, knn1, KN);
    build_group_k<<<CDIV((long)64 * Lg, 256), 256, 0, stream>>>(h2, fidx1, knn1, bigA, 64, N, S, KN, B);
    launch_gemm(l0_w1, 0, 128, bigA, 0, bigB, 0, nullptr, 128, 128, (int)Lg, 1, false, stream);
    bn(bigB, l0_g1, l0_b1, 128, (int)Lg, 1, 1);
    launch_gemm(l0_w2, 0, 128, bigB, 0, bigA, 0, nullptr, 128, 128, (int)Lg, 1, false, stream);
    bn(bigA, l0_g2, l0_b2, 128, (int)Lg, 1, 1);
    group_max_k<<<CDIV((long)B * 128 * S, 256), 256, 0, stream>>>(bigA, f0, 128, S, KN, B);

    // ============ sample_and_group + local_op #1 (D=128 -> 256ch) ============
    fps_k<<<B, 256, 0, stream>>>(nx1, S, S, fidx2);
    gather3_k<<<CDIV((long)B * S * 3, 256), 256, 0, stream>>>(nx1, fidx2, nx2, S, S, B);
    knn_k<<<dim3(S, B), 256, 0, stream>>>(nx2, nx1, S, S, knn2, KN);
    build_group_k<<<CDIV((long)128 * Lg, 256), 256, 0, stream>>>(f0, fidx2, knn2, bigA, 128, S, S, KN, B);
    launch_gemm(l1_w1, 0, 256, bigA, 0, bigB, 0, nullptr, 256, 256, (int)Lg, 1, false, stream);
    bn(bigB, l1_g1, l1_b1, 256, (int)Lg, 1, 1);
    launch_gemm(l1_w2, 0, 256, bigB, 0, bigA, 0, nullptr, 256, 256, (int)Lg, 1, false, stream);
    bn(bigA, l1_g2, l1_b2, 256, (int)Lg, 1, 1);
    group_max_k<<<CDIV((long)B * 256 * S, 256), 256, 0, stream>>>(bigA, f1, 256, S, KN, B);

    // ============ positional encoding + input embedding ============
    transpose3_k<<<CDIV((long)B * 3 * S, 256), 256, 0, stream>>>(nx2, nx2T, S, B);
    launch_gemm(pos_w, 0, 3,   nx2T, (long)3 * S,   pos, (long)128 * S, pos_b,  128, 3,   S, B, false, stream);
    launch_gemm(int_w, 0, 256, f1,   (long)256 * S, cur, (long)128 * S, nullptr, 128, 256, S, B, false, stream);
    bn(cur, int_g, int_b, 128, S, B, 1);

    // ============ 4x offset-attention layers (att lives in 192MB L2) ============
    const float* prev = cur;
    for (int i = 0; i < 4; ++i) {
        float* sa_i = saout + (long)i * B * 128 * N;
        add_k<<<CDIV(EN, 256), 256, 0, stream>>>(prev, pos, xin, EN);
        // Q (= K, shared weights): [32,128] x xin -> [B,32,N]
        launch_gemm(wqk[i], 0, 128, xin, (long)128 * N, qb, (long)32 * N, nullptr, 32, 128, N, B, false, stream);
        // Gram: att[n,m] = sum_c Q[c,n] Q[c,m]  (A column-major = Q)
        launch_gemm(qb, (long)32 * N, N, qb, (long)32 * N, bigA, (long)N * N, nullptr, N, 32, N, B, true, stream);
        softmax_row_k<<<B * N, 256, 0, stream>>>(bigA, N);
        colsum_k<<<CDIV((long)B * N, 256), 256, 0, stream>>>(bigA, cs, N, B);
        // V
        launch_gemm(wv[i], 0, 128, xin, (long)128 * N, vb, (long)128 * N, bv[i], 128, 128, N, B, false, stream);
        // xr = V x att  (A = V row-major [128][N])
        launch_gemm(vb, (long)128 * N, N, bigA, (long)N * N, xr, (long)128 * N, nullptr, 128, N, N, B, false, stream);
        colscale_k<<<CDIV(EN, 256), 256, 0, stream>>>(xr, cs, 128, N, EN);
        sub_k<<<CDIV(EN, 256), 256, 0, stream>>>(xin, xr, td, EN);
        launch_gemm(wt[i], 0, 128, td, (long)128 * N, ub, (long)128 * N, bt[i], 128, 128, N, B, false, stream);
        bn(ub, sag[i], sab[i], 128, N, B, 1);
        add_k<<<CDIV(EN, 256), 256, 0, stream>>>(xin, ub, sa_i, EN);
        prev = sa_i;
    }

    // ============ fuse + global features ============
    concat4_k<<<CDIV((long)B * 512 * N, 256), 256, 0, stream>>>(
        saout, saout + (long)B * 128 * N, saout + (long)2 * B * 128 * N,
        saout + (long)3 * B * 128 * N, xc, 128, N, (long)B * 512 * N);
    launch_gemm(fuse_w, 0, 512, xc, (long)512 * N, xf, (long)1024 * N, nullptr, 1024, 512, N, B, false, stream);
    bn(xf, fuse_g, fuse_b, 1024, N, B, 2);
    maxavg_k<<<B * 1024, 256, 0, stream>>>(xf, xmaxb, xavgb, N);
    launch_gemm(lab_w, 0, 16, clsl, 16, lf, 64, nullptr, 64, 16, 1, B, false, stream);
    bn(lf, lab_g, lab_b, 64, 1, B, 2);
    build_y_k<<<CDIV((long)B * 3136 * N, 256), 256, 0, stream>>>(
        xf, xmaxb, xavgb, lf, bigA, N, (long)B * 3136 * N);

    // ============ segmentation head ============
    launch_gemm(s1_w, 0, 3136, bigA, (long)3136 * N, y1, (long)512 * N, s1_bias, 512, 3136, N, B, false, stream);
    bn(y1, s1_g, s1_b, 512, N, B, 1);
    launch_gemm(s2_w, 0, 512, y1, (long)512 * N, y2, (long)256 * N, s2_bias, 256, 512, N, B, false, stream);
    bn(y2, s2_g, s2_b, 256, N, B, 1);
    launch_gemm(s3_w, 0, 256, y2, (long)256 * N, (float*)d_out, (long)50 * N, s3_bias, 50, 256, N, B, false, stream);
}
